// EnhancedTFNLayer_6914897346624
// MI455X (gfx1250) — matrix operand order repaired and checked by reading, not compiled, not run on hardware
//
#include <hip/hip_runtime.h>

#define Bq 8
#define Nq 2048
#define Dq 512
#define Gq 1024
#define MAXLENq 2048
#define NSTEPS 4

typedef __attribute__((ext_vector_type(16))) __bf16 v16bf;
typedef __attribute__((ext_vector_type(8)))  __bf16 v8bf;
typedef __attribute__((ext_vector_type(8)))  float  v8f;
typedef __attribute__((ext_vector_type(4)))  unsigned int v4u;
typedef __attribute__((ext_vector_type(8)))  int v8i;
typedef __attribute__((ext_vector_type(4)))  int v4i;

#if defined(__has_builtin)
#if __has_builtin(__builtin_amdgcn_tensor_load_to_lds) && \
    __has_builtin(__builtin_amdgcn_s_wait_tensorcnt)
#define USE_TDM 1
#endif
#endif

// ---------------------------------------------------------------------------
// WMMA fragment helpers (layouts per CDNA5 ISA §7.12.2)
// A 16x32 bf16, lane L: row m=L&15, hi=L>>4:
//   vector elems 0..7  = K = kb + hi*8 + 0..7
//   vector elems 8..15 = K = kb + 16 + hi*8 + 0..7
static __device__ __forceinline__ v16bf load_a_frag(const __bf16* __restrict__ row,
                                                    int kb, int hi) {
  const v8bf lo = *reinterpret_cast<const v8bf*>(row + kb + hi * 8);
  const v8bf hh = *reinterpret_cast<const v8bf*>(row + kb + 16 + hi * 8);
  v16bf a;
#pragma unroll
  for (int i = 0; i < 8; ++i) { a[i] = lo[i]; a[8 + i] = hh[i]; }
  return a;
}

// B 32x16 bf16, lane L: col n=L&15, hi=L>>4; elems 0..7 = K = kb+hi*8+0..7,
// elems 8..15 = K = kb+16+hi*8+0..7 (same K order as A).
static __device__ __forceinline__ v16bf load_b_frag(const __bf16* __restrict__ krow,
                                                    int kb, int hi) {
  const v8bf lo = *reinterpret_cast<const v8bf*>(krow + kb + hi * 8);
  const v8bf hh = *reinterpret_cast<const v8bf*>(krow + kb + 16 + hi * 8);
  v16bf b;
#pragma unroll
  for (int i = 0; i < 8; ++i) { b[i] = lo[i]; b[8 + i] = hh[i]; }
  return b;
}

static __device__ __forceinline__ v8f wmma_bf16(v16bf a, v16bf b, v8f c) {
  return __builtin_amdgcn_wmma_f32_16x16x32_bf16(false, a, false, b, (short)0, c,
                                                 false, false);
}

static __device__ __forceinline__ void reduce2_128(float& s, float& s2,
                                                   float* r1, float* r2) {
  const int t = threadIdx.x;
  r1[t] = s; r2[t] = s2; __syncthreads();
  for (int stp = 64; stp > 0; stp >>= 1) {
    if (t < stp) { r1[t] += r1[t + stp]; r2[t] += r2[t + stp]; }
    __syncthreads();
  }
  s = r1[0]; s2 = r2[0]; __syncthreads();
}

#ifdef USE_TDM
// One 1-D TDM tile: copy nelem f32 from global row -> LDS (wave-level op).
// 6-arg builtin form (amdgpu-toolchain / clang-23):
//   (uint32x4 g0, int32x8 g1, int32x4 g2, int32x4 g3, int32x8 g4, i32 cpol)
static __device__ __forceinline__ void tdm_load_row_f32(const float* gsrc,
                                                        unsigned int lds_off,
                                                        int nelem) {
  const unsigned long long ga = (unsigned long long)gsrc;
  v4u g0;
  g0[0] = 1u;                                        // count=1, user desc
  g0[1] = lds_off;                                   // lds_addr (bytes)
  g0[2] = (unsigned int)ga;                          // global_addr[31:0]
  g0[3] = (unsigned int)(ga >> 32) | (2u << 30);     // addr[56:32] | type=2
  v8i g1;
  g1[0] = (int)(2u << 16);                 // data_size=4B, no mask/pad/iter
  g1[1] = (int)((unsigned)nelem << 16);    // tensor_dim0[15:0]
  g1[2] = (int)(1u << 16);                 // tensor_dim1 = 1
  g1[3] = (int)((unsigned)nelem << 16);    // tile_dim0
  g1[4] = 0;                               // tile_dim1/2 unused
  g1[5] = nelem;                           // tensor_dim0_stride[31:0]
  g1[6] = 0;
  g1[7] = 0;
  v4i g2 = {0, 0, 0, 0};
  v4i g3 = {0, 0, 0, 0};
  v8i g4 = {0, 0, 0, 0, 0, 0, 0, 0};
  __builtin_amdgcn_tensor_load_to_lds(g0, g1, g2, g3, g4, 0);
}
#endif

// ---------------------------------------------------------------------------
// 1) emb = embeddings + pos_table[round(pos*(L-1))]; f32 + bf16 copies
__global__ void __launch_bounds__(128) emb_kernel(
    const float* __restrict__ embeddings, const float* __restrict__ positions,
    const float* __restrict__ pos_table, float* __restrict__ embf,
    __bf16* __restrict__ embb) {
  const int tok = blockIdx.x;
  const float p = positions[tok];
  int idx = __float2int_rn(p * (float)(MAXLENq - 1));
  idx = idx < 0 ? 0 : (idx > MAXLENq - 1 ? MAXLENq - 1 : idx);
  const float* e = embeddings + (size_t)tok * Dq;
  const float* t = pos_table + (size_t)idx * Dq;
  float* of = embf + (size_t)tok * Dq;
  __bf16* ob = embb + (size_t)tok * Dq;
#pragma unroll
  for (int k = 0; k < 4; ++k) {
    const int d = threadIdx.x + 128 * k;
    const float v = e[d] + t[d];
    of[d] = v;
    ob[d] = (__bf16)v;
  }
}

__global__ void f2bf_kernel(const float* __restrict__ s, __bf16* __restrict__ d,
                            int n) {
  const int i = blockIdx.x * 256 + threadIdx.x;
  if (i < n) d[i] = (__bf16)s[i];
}

// batched 32x32 tiled transpose: src[b][R][C] -> dst[b][C][R]  (bf16)
__global__ void __launch_bounds__(256) transpose_bf16_kernel(
    const __bf16* __restrict__ src, __bf16* __restrict__ dst, int R, int C) {
  __shared__ __bf16 t[32][33];
  const size_t base = (size_t)blockIdx.z * R * C;
  const int r0 = blockIdx.y << 5, c0 = blockIdx.x << 5;
  const int tx = threadIdx.x & 31, ty = threadIdx.x >> 5;
#pragma unroll
  for (int k = 0; k < 4; ++k)
    t[ty + 8 * k][tx] = src[base + (size_t)(r0 + ty + 8 * k) * C + c0 + tx];
  __syncthreads();
#pragma unroll
  for (int k = 0; k < 4; ++k)
    dst[base + (size_t)(c0 + ty + 8 * k) * R + r0 + tx] = t[tx][ty + 8 * k];
}

// 32x32 tiled transpose with f32 -> bf16 convert (for w_out^T)
__global__ void __launch_bounds__(256) transpose_f2bf_kernel(
    const float* __restrict__ src, __bf16* __restrict__ dst, int R, int C) {
  __shared__ __bf16 t[32][33];
  const int r0 = blockIdx.y << 5, c0 = blockIdx.x << 5;
  const int tx = threadIdx.x & 31, ty = threadIdx.x >> 5;
#pragma unroll
  for (int k = 0; k < 4; ++k)
    t[ty + 8 * k][tx] = (__bf16)src[(size_t)(r0 + ty + 8 * k) * C + c0 + tx];
  __syncthreads();
#pragma unroll
  for (int k = 0; k < 4; ++k)
    dst[(size_t)(c0 + ty + 8 * k) * R + r0 + tx] = t[tx][ty + 8 * k];
}

// ---------------------------------------------------------------------------
// 2) field[b,g,d] = sum_n exp(-(grid_g-pos_n)^2/(2s^2)) * emb[b,n,d]
//    A (16g x 32n) on the fly from LDS positions (TDM-staged); B from embT.
//    Wave computes 16x64; block 128 = 4 waves -> 16g x 256d.
__global__ void __launch_bounds__(128) project_kernel(
    const float* __restrict__ positions, const float* __restrict__ grid_points,
    const float* __restrict__ sigma, const __bf16* __restrict__ embT,
    float* __restrict__ field) {
  const int b = blockIdx.z;
  const int g0 = blockIdx.y << 4;
  const int lane = threadIdx.x & 31, wv = threadIdx.x >> 5;
  const int hi = lane >> 4, ln = lane & 15;
  const int d0 = (blockIdx.x << 8) + (wv << 6);

  __shared__ __align__(16) float sPos[Nq];

#ifdef USE_TDM
  if (threadIdx.x == 0) {
    tdm_load_row_f32(positions + (size_t)b * Nq,
                     (unsigned int)(unsigned long long)&sPos[0], Nq);
    __builtin_amdgcn_s_wait_tensorcnt((short)0);
  }
  __syncthreads();
#else
  for (int i = threadIdx.x; i < Nq; i += 128)
    sPos[i] = positions[b * Nq + i];
  __syncthreads();
#endif

  const float sg = sigma[0];
  const float i2s = 1.0f / (2.0f * sg * sg);
  const float gp = grid_points[b * Gq + g0 + ln];

  const __bf16* tb = embT + (size_t)b * Dq * Nq;
  const __bf16* bp[4];
#pragma unroll
  for (int j = 0; j < 4; ++j) bp[j] = tb + (size_t)(d0 + (j << 4) + ln) * Nq;

  v8f cc[4] = {};
  for (int kb = 0; kb < Nq; kb += 32) {
    v16bf a;
#pragma unroll
    for (int i = 0; i < 8; ++i) {
      const int km = (i < 4 ? 2 * i : 16 + 2 * (i - 4)) + hi * 8;
      const float2 pp = *reinterpret_cast<const float2*>(&sPos[kb + km]);
      const float dd0 = gp - pp.x;
      const float dd1 = gp - pp.y;
      a[2 * i]     = (__bf16)__expf(-dd0 * dd0 * i2s);
      a[2 * i + 1] = (__bf16)__expf(-dd1 * dd1 * i2s);
    }
#pragma unroll
    for (int j = 0; j < 4; ++j)
      cc[j] = wmma_bf16(a, load_b_frag(bp[j], kb, hi), cc[j]);
  }
#pragma unroll
  for (int j = 0; j < 4; ++j) {
    const int dc = d0 + (j << 4) + ln;
#pragma unroll
    for (int r = 0; r < 8; ++r)
      field[((size_t)b * Gq + g0 + r + hi * 8) * Dq + dc] = cc[j][r];
  }
}

// ---------------------------------------------------------------------------
// 3a) scores = F F^T, both operands read row-major (K = d, contiguous).
//     Wave computes 16g x 64h; block 256 = 8 waves -> 16g x 512h.
__global__ void __launch_bounds__(256) scores_kernel(
    const __bf16* __restrict__ fb, float* __restrict__ scores) {
  const int b = blockIdx.z;
  const int g0 = blockIdx.y << 4;
  const int lane = threadIdx.x & 31, wv = threadIdx.x >> 5;
  const int hi = lane >> 4, ln = lane & 15;
  const int h0 = (blockIdx.x << 9) + (wv << 6);

  const __bf16* base = fb + (size_t)b * Gq * Dq;
  const __bf16* aRow = base + (size_t)(g0 + ln) * Dq;
  const __bf16* bp[4];
#pragma unroll
  for (int j = 0; j < 4; ++j) bp[j] = base + (size_t)(h0 + (j << 4) + ln) * Dq;

  v8f cc[4] = {};
  for (int kb = 0; kb < Dq; kb += 32) {
    if (kb + 32 < Dq) __builtin_prefetch(aRow + kb + 32, 0, 0);
    const v16bf a = load_a_frag(aRow, kb, hi);
#pragma unroll
    for (int j = 0; j < 4; ++j)
      cc[j] = wmma_bf16(a, load_b_frag(bp[j], kb, hi), cc[j]);
  }
  float* srow = scores + (size_t)b * Gq * Gq;
#pragma unroll
  for (int r = 0; r < 8; ++r) {
    const size_t ro = (size_t)(g0 + r + hi * 8) * Gq + h0 + ln;
#pragma unroll
    for (int j = 0; j < 4; ++j) srow[ro + (j << 4)] = cc[j][r];
  }
}

// 3b) row softmax (with 1/sqrt(D) scale), bf16 attn out
__global__ void __launch_bounds__(256) softmax_kernel(
    const float* __restrict__ scores, __bf16* __restrict__ attnb) {
  const int row = blockIdx.x;
  const float scale = 0.044194173824159216f;  // 1/sqrt(512)
  const float* s = scores + (size_t)row * Gq;
  __bf16* o = attnb + (size_t)row * Gq;
  __shared__ float red[256];
  const int t = threadIdx.x;

  float v[4];
  float m = -3.0e38f;
#pragma unroll
  for (int k = 0; k < 4; ++k) {
    v[k] = s[t + (k << 8)] * scale;
    m = v[k] > m ? v[k] : m;
  }
  red[t] = m; __syncthreads();
  for (int stp = 128; stp > 0; stp >>= 1) {
    if (t < stp) red[t] = red[t] > red[t + stp] ? red[t] : red[t + stp];
    __syncthreads();
  }
  m = red[0]; __syncthreads();

  float sum = 0.f;
#pragma unroll
  for (int k = 0; k < 4; ++k) { v[k] = __expf(v[k] - m); sum += v[k]; }
  red[t] = sum; __syncthreads();
  for (int stp = 128; stp > 0; stp >>= 1) {
    if (t < stp) red[t] += red[t + stp];
    __syncthreads();
  }
  const float inv = 1.0f / red[0];
#pragma unroll
  for (int k = 0; k < 4; ++k) o[t + (k << 8)] = (__bf16)(v[k] * inv);
}

// 3c) inter = attn @ F (A row-major attn, B from field^T), fused diffusion
//     update; writes f32 field', row-major bf16, and K-major bf16.
__global__ void __launch_bounds__(128) inter_update_kernel(
    const __bf16* __restrict__ attnb, const __bf16* __restrict__ fbT,
    const float* __restrict__ fOld, const float* __restrict__ alpha,
    const float* __restrict__ beta_p, const float* __restrict__ dt_p,
    float* __restrict__ fNew, __bf16* __restrict__ fbNew,
    __bf16* __restrict__ fbTNew) {
  const int b = blockIdx.z;
  const int g0 = blockIdx.y << 4;
  const int lane = threadIdx.x & 31, wv = threadIdx.x >> 5;
  const int hi = lane >> 4, ln = lane & 15;
  const int d0 = (blockIdx.x << 8) + (wv << 6);

  const __bf16* aRow = attnb + ((size_t)b * Gq + g0 + ln) * Gq;
  const __bf16* tb = fbT + (size_t)b * Dq * Gq;
  const __bf16* bp[4];
#pragma unroll
  for (int j = 0; j < 4; ++j) bp[j] = tb + (size_t)(d0 + (j << 4) + ln) * Gq;

  v8f cc[4] = {};
  for (int kb = 0; kb < Gq; kb += 32) {
    if (kb + 32 < Gq) __builtin_prefetch(aRow + kb + 32, 0, 0);
    const v16bf a = load_a_frag(aRow, kb, hi);
#pragma unroll
    for (int j = 0; j < 4; ++j)
      cc[j] = wmma_bf16(a, load_b_frag(bp[j], kb, hi), cc[j]);
  }

  const float beta = beta_p[0];
  const float dt = dt_p[0];
  const size_t bo = (size_t)b * Gq * Dq;
  const float* fB = fOld + bo;
#pragma unroll
  for (int j = 0; j < 4; ++j) {
    const int dc = d0 + (j << 4) + ln;
    const float al = alpha[dc];
    v8bf pk;
#pragma unroll
    for (int r = 0; r < 8; ++r) {
      const int g = g0 + r + hi * 8;
      const size_t idx = (size_t)g * Dq + dc;
      const float f = fB[idx];
      const int gm = g > 0 ? g - 1 : 0;
      const int gp2 = g < Gq - 1 ? g + 1 : Gq - 1;
      const float lap =
          fB[(size_t)gm * Dq + dc] + fB[(size_t)gp2 * Dq + dc] - 2.0f * f;
      const float nv = f + dt * (al * lap + beta * cc[j][r]);
      fNew[bo + idx] = nv;
      fbNew[bo + idx] = (__bf16)nv;
      pk[r] = (__bf16)nv;
    }
    *reinterpret_cast<v8bf*>(fbTNew + ((size_t)b * Dq + dc) * Gq + g0 + hi * 8) = pk;
  }
}

// ---------------------------------------------------------------------------
// 4) sample field at token positions, residual + LN1; f32 + bf16 out
__global__ void __launch_bounds__(128) sample_ln1_kernel(
    const float* __restrict__ positions, const float* __restrict__ field,
    const float* __restrict__ embf, const float* __restrict__ g1,
    const float* __restrict__ b1, float* __restrict__ enh,
    __bf16* __restrict__ enhb) {
  __shared__ float r1[128], r2[128];
  const int tok = blockIdx.x;
  const int b = tok / Nq;
  const float u = positions[tok] * (float)(Gq - 1);
  int i0 = (int)floorf(u);
  i0 = i0 < 0 ? 0 : (i0 > Gq - 1 ? Gq - 1 : i0);
  const int i1 = i0 + 1 < Gq ? i0 + 1 : Gq - 1;
  const float w = u - (float)i0;
  const float* f0 = field + ((size_t)b * Gq + i0) * Dq;
  const float* f1 = field + ((size_t)b * Gq + i1) * Dq;
  const float* e = embf + (size_t)tok * Dq;

  float x[4];
  float s = 0.f, s2 = 0.f;
#pragma unroll
  for (int k = 0; k < 4; ++k) {
    const int d = threadIdx.x + 128 * k;
    const float v = f0[d] * (1.0f - w) + f1[d] * w + e[d];
    x[k] = v; s += v; s2 += v * v;
  }
  reduce2_128(s, s2, r1, r2);
  const float mean = s * (1.0f / Dq);
  const float var = s2 * (1.0f / Dq) - mean * mean;
  const float rs = rsqrtf(var + 1e-5f);
#pragma unroll
  for (int k = 0; k < 4; ++k) {
    const int d = threadIdx.x + 128 * k;
    const float y = (x[k] - mean) * rs * g1[d] + b1[d];
    enh[(size_t)tok * Dq + d] = y;
    enhb[(size_t)tok * Dq + d] = (__bf16)y;
  }
}

// ---------------------------------------------------------------------------
// 5) out-projection: tmp = enhanced @ w_out + b_out (B from w_out^T, K-major)
//    Wave 16tok x 64dout; block 256 = 8 waves -> full 512 dout.
__global__ void __launch_bounds__(256) outproj_kernel(
    const __bf16* __restrict__ enhb, const __bf16* __restrict__ wT,
    const float* __restrict__ b_out, float* __restrict__ tmp) {
  const int t0 = blockIdx.x << 4;
  const int lane = threadIdx.x & 31, wv = threadIdx.x >> 5;
  const int hi = lane >> 4, ln = lane & 15;
  const int d0 = wv << 6;

  const __bf16* aRow = enhb + (size_t)(t0 + ln) * Dq;
  const __bf16* bp[4];
#pragma unroll
  for (int j = 0; j < 4; ++j) bp[j] = wT + (size_t)(d0 + (j << 4) + ln) * Dq;

  v8f cc[4] = {};
  for (int kb = 0; kb < Dq; kb += 32) {
    const v16bf a = load_a_frag(aRow, kb, hi);
#pragma unroll
    for (int j = 0; j < 4; ++j)
      cc[j] = wmma_bf16(a, load_b_frag(bp[j], kb, hi), cc[j]);
  }
#pragma unroll
  for (int j = 0; j < 4; ++j) {
    const int dc = d0 + (j << 4) + ln;
    const float bo = b_out[dc];
#pragma unroll
    for (int r = 0; r < 8; ++r)
      tmp[(size_t)(t0 + r + hi * 8) * Dq + dc] = cc[j][r] + bo;
  }
}

// 6) final LN2 over (tmp + enhanced) -> d_out
__global__ void __launch_bounds__(128) ln2_kernel(
    const float* __restrict__ tmp, const float* __restrict__ enh,
    const float* __restrict__ g2, const float* __restrict__ b2,
    float* __restrict__ out) {
  __shared__ float r1[128], r2[128];
  const int tok = blockIdx.x;
  const float* a = tmp + (size_t)tok * Dq;
  const float* e = enh + (size_t)tok * Dq;

  float x[4];
  float s = 0.f, s2 = 0.f;
#pragma unroll
  for (int k = 0; k < 4; ++k) {
    const int d = threadIdx.x + 128 * k;
    const float v = a[d] + e[d];
    x[k] = v; s += v; s2 += v * v;
  }
  reduce2_128(s, s2, r1, r2);
  const float mean = s * (1.0f / Dq);
  const float var = s2 * (1.0f / Dq) - mean * mean;
  const float rs = rsqrtf(var + 1e-5f);
#pragma unroll
  for (int k = 0; k < 4; ++k) {
    const int d = threadIdx.x + 128 * k;
    out[(size_t)tok * Dq + d] = (x[k] - mean) * rs * g2[d] + b2[d];
  }
}

// ---------------------------------------------------------------------------
extern "C" void kernel_launch(void* const* d_in, const int* in_sizes, int n_in,
                              void* d_out, int out_size, void* d_ws,
                              size_t ws_size, hipStream_t stream) {
  (void)in_sizes; (void)n_in; (void)out_size; (void)ws_size;
  const float* embeddings = (const float*)d_in[0];
  const float* positions  = (const float*)d_in[1];
  const float* grid_pts   = (const float*)d_in[2];
  const float* pos_table  = (const float*)d_in[3];
  const float* sigma      = (const float*)d_in[4];
  const float* alpha      = (const float*)d_in[5];
  const float* beta       = (const float*)d_in[6];
  const float* dt         = (const float*)d_in[7];
  const float* ln1_g      = (const float*)d_in[8];
  const float* ln1_b      = (const float*)d_in[9];
  const float* ln2_g      = (const float*)d_in[10];
  const float* ln2_b      = (const float*)d_in[11];
  const float* w_out      = (const float*)d_in[12];
  const float* b_out      = (const float*)d_in[13];

  const size_t BND = (size_t)Bq * Nq * Dq;  // 8,388,608
  const size_t BGD = (size_t)Bq * Gq * Dq;  // 4,194,304
  const size_t BGG = (size_t)Bq * Gq * Gq;  // 8,388,608

  char* ws = (char*)d_ws;
  size_t off = 0;
  auto alloc = [&](size_t bytes) -> void* {
    void* p = ws + off;
    off += (bytes + 255) & ~(size_t)255;
    return p;
  };
  float*  embf   = (float*) alloc(BND * 4);
  __bf16* embb   = (__bf16*)alloc(BND * 2);
  __bf16* embT   = (__bf16*)alloc(BND * 2);
  float*  fieldA = (float*) alloc(BGD * 4);
  float*  fieldB = (float*) alloc(BGD * 4);
  __bf16* fbA    = (__bf16*)alloc(BGD * 2);
  __bf16* fbB    = (__bf16*)alloc(BGD * 2);
  __bf16* fbTA   = (__bf16*)alloc(BGD * 2);
  __bf16* fbTB   = (__bf16*)alloc(BGD * 2);
  float*  scores = (float*) alloc(BGG * 4);
  __bf16* attnb  = (__bf16*)alloc(BGG * 2);
  float*  enh    = (float*) alloc(BND * 4);
  __bf16* enhb   = (__bf16*)alloc(BND * 2);
  __bf16* woutT  = (__bf16*)alloc((size_t)Dq * Dq * 2);
  float*  tmp    = scores;  // scores buffer reused for out-projection

  emb_kernel<<<Bq * Nq, 128, 0, stream>>>(embeddings, positions, pos_table,
                                          embf, embb);
  transpose_bf16_kernel<<<dim3(Dq / 32, Nq / 32, Bq), 256, 0, stream>>>(
      embb, embT, Nq, Dq);
  transpose_f2bf_kernel<<<dim3(Dq / 32, Dq / 32, 1), 256, 0, stream>>>(
      w_out, woutT, Dq, Dq);

  project_kernel<<<dim3(2, Gq / 16, Bq), 128, 0, stream>>>(
      positions, grid_pts, sigma, embT, fieldA);
  f2bf_kernel<<<(int)((BGD + 255) / 256), 256, 0, stream>>>(fieldA, fbA,
                                                            (int)BGD);
  transpose_bf16_kernel<<<dim3(Dq / 32, Gq / 32, Bq), 256, 0, stream>>>(
      fbA, fbTA, Gq, Dq);

  float*  fc = fieldA;  __bf16* fbc = fbA;  __bf16* fbTc = fbTA;
  float*  fn = fieldB;  __bf16* fbn = fbB;  __bf16* fbTn = fbTB;
  for (int s = 0; s < NSTEPS; ++s) {
    scores_kernel<<<dim3(2, Gq / 16, Bq), 256, 0, stream>>>(fbc, scores);
    softmax_kernel<<<Bq * Gq, 256, 0, stream>>>(scores, attnb);
    inter_update_kernel<<<dim3(2, Gq / 16, Bq), 128, 0, stream>>>(
        attnb, fbTc, fc, alpha, beta, dt, fn, fbn, fbTn);
    float* tf = fc; fc = fn; fn = tf;
    __bf16* tb1 = fbc; fbc = fbn; fbn = tb1;
    __bf16* tb2 = fbTc; fbTc = fbTn; fbTn = tb2;
  }

  sample_ln1_kernel<<<Bq * Nq, 128, 0, stream>>>(positions, fc, embf, ln1_g,
                                                 ln1_b, enh, enhb);
  outproj_kernel<<<(Bq * Nq) / 16, 256, 0, stream>>>(enhb, woutT, b_out, tmp);
  ln2_kernel<<<Bq * Nq, 128, 0, stream>>>(tmp, enh, ln2_g, ln2_b,
                                          (float*)d_out);
}
